// NeuralHMM_8186207666595
// MI455X (gfx1250) — compile-verified, matrix operand order checked by reading, not run on hardware
//
#include <hip/hip_runtime.h>
#include <hip/hip_bf16.h>

#define BB 32      // batch
#define TT 512     // max sequence length
#define HH 64      // hidden states
#define OO 64      // observation symbols

typedef __attribute__((ext_vector_type(16))) _Float16 v16h;
typedef __attribute__((ext_vector_type(8)))  float    v8f;

#define NEG_INF (-__builtin_inff())

// ---------------------------------------------------------------------------
// WMMA helper: D = A(16x32 f16) * B(32x16 f16) + C(16x16 f32)
// ---------------------------------------------------------------------------
__device__ __forceinline__ v8f wmma_f16f32(v16h a, v16h b, v8f c) {
  return __builtin_amdgcn_wmma_f32_16x16x32_f16(
      /*neg_a=*/false, a, /*neg_b=*/false, b,
      /*c_mod=*/(short)0, c, /*reuse_a=*/false, /*reuse_b=*/false);
}

// A fragment: 16x32 f16 tile from a row-major [M][K] source (ld = row stride).
// Lane layout per CDNA5 ISA 7.12.2 (16-bit A-matrix 16x32):
//   lanes 0-15: M=lane, kbase=0; lanes 16-31: M=lane-16, kbase=8
//   VGPR v in 0..3 holds K = kbase+2v, +1 ; v in 4..7 holds K = 16+kbase+2(v-4), +1
__device__ __forceinline__ v16h load_afrag(const _Float16* src, int row,
                                           int kbase, int k0, int ld) {
  v16h f;
#pragma unroll
  for (int v = 0; v < 8; ++v) {
    int kk = ((v < 4) ? (kbase + 2 * v) : (16 + kbase + 2 * (v - 4))) + k0;
    f[2 * v]     = src[row * ld + kk];
    f[2 * v + 1] = src[row * ld + kk + 1];
  }
  return f;
}

// B fragment: 32x16 f16 tile (KxN) from a row-major [K][N] source (ld = row
// stride). Mirrors the A layout with M<->N.
__device__ __forceinline__ v16h load_bfrag(const _Float16* src, int col,
                                           int kbase, int k0, int ld) {
  v16h f;
#pragma unroll
  for (int v = 0; v < 8; ++v) {
    int kk = ((v < 4) ? (kbase + 2 * v) : (16 + kbase + 2 * (v - 4))) + k0;
    f[2 * v]     = src[kk * ld + col];
    f[2 * v + 1] = src[(kk + 1) * ld + col];
  }
  return f;
}

// ---------------------------------------------------------------------------
// K0: parameter log-softmax + exp(transition) in f16 (both orientations).
// Also zeroes the scalar output (stream-ordered before the atomics).
// ---------------------------------------------------------------------------
__global__ void hmm_prep(const float* __restrict__ sp,
                         const float* __restrict__ utr,
                         const float* __restrict__ uem,
                         float* __restrict__ logPriors,
                         float* __restrict__ lT,
                         float* __restrict__ lE,
                         _Float16* __restrict__ expT,
                         _Float16* __restrict__ expTT,
                         float* __restrict__ out) {
  const int tid = threadIdx.x;  // 128 threads
  if (tid == 0) {
    out[0] = 0.f;
    float mx = NEG_INF;
    for (int h = 0; h < HH; ++h) mx = fmaxf(mx, sp[h]);
    float s = 0.f;
    for (int h = 0; h < HH; ++h) s += __expf(sp[h] - mx);
    float lse = mx + __logf(s);
    for (int h = 0; h < HH; ++h) logPriors[h] = sp[h] - lse;
  }
  if (tid < HH) {  // transition row tid
    const int r = tid;
    float mx = NEG_INF;
    for (int c = 0; c < HH; ++c) mx = fmaxf(mx, utr[r * HH + c]);
    float s = 0.f;
    for (int c = 0; c < HH; ++c) s += __expf(utr[r * HH + c] - mx);
    float lse = mx + __logf(s);
    for (int c = 0; c < HH; ++c) {
      float v = utr[r * HH + c] - lse;
      lT[r * HH + c] = v;
      float e = __expf(v);
      expT[r * HH + c]  = (_Float16)e;   // [from][to]
      expTT[c * HH + r] = (_Float16)e;   // [to][from]
    }
  }
  if (tid >= 64 && tid < 64 + HH) {  // emission row tid-64
    const int r = tid - 64;
    float mx = NEG_INF;
    for (int c = 0; c < OO; ++c) mx = fmaxf(mx, uem[r * OO + c]);
    float s = 0.f;
    for (int c = 0; c < OO; ++c) s += __expf(uem[r * OO + c] - mx);
    float lse = mx + __logf(s);
    for (int c = 0; c < OO; ++c) lE[r * OO + c] = uem[r * OO + c] - lse;
  }
}

// ---------------------------------------------------------------------------
// K1: one-hot gather -> lpx[t][b][h] = log_emiss[h, idx(b,t)]   (time-major)
// ---------------------------------------------------------------------------
__global__ void hmm_lpx(const float* __restrict__ obs,
                        const float* __restrict__ lE,
                        float* __restrict__ lpx) {
  const int bt = blockIdx.x;        // B*T blocks
  const int b = bt / TT, t = bt % TT;
  const int h = threadIdx.x;        // 64 threads
  __shared__ int sidx;
  if (h == 0) sidx = 0;
  __syncthreads();
  if (obs[(size_t)(b * TT + t) * OO + h] > 0.5f) sidx = h;
  __syncthreads();
  lpx[((size_t)t * BB + b) * HH + h] = lE[h * OO + sidx];
}

// ---------------------------------------------------------------------------
// K2: persistent recurrence. Block 0 = forward, Block 1 = backward (they are
// independent given lpx). State lives in LDS; per step: max-shift row phase,
// then 8 waves x 2 v_wmma_f32_16x16x32_f16 (K=64), log() back to LDS + HBM.
// ---------------------------------------------------------------------------
__global__ __launch_bounds__(256) void hmm_recurrence(
    const float* __restrict__ lpx,        // [T][B][H]
    const float* __restrict__ logPriors,  // [H]
    const _Float16* __restrict__ expT,    // [H][H]
    const _Float16* __restrict__ expTT,   // [H][H] transposed
    float* __restrict__ alpha_g,          // [T][B][H]
    float* __restrict__ beta_g) {         // [T][B][H]
  __shared__ float    sA[BB * HH];   // log-domain staging (alpha pre-norm / beta)
  __shared__ _Float16 sP[BB * HH];   // exp-shifted operand for WMMA
  __shared__ float    sM[BB];        // per-row shift

  const int tid  = threadIdx.x;
  const int lane = tid & 31;
  const int wave = tid >> 5;
  const int ib = wave >> 2;              // batch-row block (0..1)
  const int jb = wave & 3;               // state-col block (0..3)
  const int nloc  = lane & 15;
  const int kbase = (lane < 16) ? 0 : 8;
  const int rbase = (lane < 16) ? 0 : 8;
  const int brow = ib * 16 + nloc;       // A-matrix row for this lane
  const int ncol = jb * 16 + nloc;       // C/D column for this lane

  if (blockIdx.x == 0) {
    // ------------------------- forward -------------------------
    v16h bf0 = load_bfrag(expT, ncol, kbase, 0,  HH);
    v16h bf1 = load_bfrag(expT, ncol, kbase, 32, HH);

    for (int i = tid; i < BB * HH; i += 256)
      sA[i] = logPriors[i & (HH - 1)] + lpx[i];   // t = 0
    __syncthreads();
    if (tid < BB) {
      const int b = tid;
      float mx = NEG_INF;
      for (int h = 0; h < HH; ++h) mx = fmaxf(mx, sA[b * HH + h]);
      float s = 0.f;
      for (int h = 0; h < HH; ++h) s += __expf(sA[b * HH + h] - mx);
      float lse = mx + __logf(s);
      float m = mx - lse;
      float* ag = alpha_g + (size_t)b * HH;
      for (int h = 0; h < HH; ++h) {
        float a = sA[b * HH + h] - lse;
        ag[h] = a;
        sP[b * HH + h] = (_Float16)__expf(a - m);
      }
      sM[b] = m;
    }
    __syncthreads();

    for (int t = 1; t < TT; ++t) {
      const float* lpx_t = lpx + (size_t)t * BB * HH;
      if (t + 1 < TT)
        __builtin_prefetch(lpx + (size_t)(t + 1) * BB * HH + tid * 8, 0, 1);
      v16h a0 = load_afrag(sP, brow, kbase, 0,  HH);
      v16h a1 = load_afrag(sP, brow, kbase, 32, HH);
      v8f c = {};
      c = wmma_f16f32(a0, bf0, c);
      c = wmma_f16f32(a1, bf1, c);
#pragma unroll
      for (int r = 0; r < 8; ++r) {
        int b = ib * 16 + rbase + r;
        sA[b * HH + ncol] = lpx_t[b * HH + ncol] + sM[b] + __logf(c[r]);
      }
      __syncthreads();
      if (tid < BB) {
        const int b = tid;
        float mx = NEG_INF;
        for (int h = 0; h < HH; ++h) mx = fmaxf(mx, sA[b * HH + h]);
        float s = 0.f;
        for (int h = 0; h < HH; ++h) s += __expf(sA[b * HH + h] - mx);
        float lse = mx + __logf(s);
        float m = mx - lse;
        float* ag = alpha_g + (size_t)t * BB * HH + (size_t)b * HH;
        for (int h = 0; h < HH; ++h) {
          float a = sA[b * HH + h] - lse;
          ag[h] = a;
          sP[b * HH + h] = (_Float16)__expf(a - m);
        }
        sM[b] = m;
      }
      __syncthreads();
    }
  } else {
    // ------------------------- backward ------------------------
    v16h bf0 = load_bfrag(expTT, ncol, kbase, 0,  HH);
    v16h bf1 = load_bfrag(expTT, ncol, kbase, 32, HH);

    for (int i = tid; i < BB * HH; i += 256) {
      sA[i] = 0.f;                                     // beta[T-1] = 0
      beta_g[(size_t)(TT - 1) * BB * HH + i] = 0.f;
    }
    __syncthreads();

    for (int t = TT - 2; t >= 0; --t) {
      const float* lpx_t = lpx + (size_t)t * BB * HH;  // reference uses lpx at t
      if (t > 0)
        __builtin_prefetch(lpx + (size_t)(t - 1) * BB * HH + tid * 8, 0, 1);
      if (tid < BB) {
        const int b = tid;
        float mx = NEG_INF;
        for (int h = 0; h < HH; ++h)
          mx = fmaxf(mx, lpx_t[b * HH + h] + sA[b * HH + h]);
        for (int h = 0; h < HH; ++h)
          sP[b * HH + h] = (_Float16)__expf(lpx_t[b * HH + h] + sA[b * HH + h] - mx);
        sM[b] = mx;                                    // no normalization (ref)
      }
      __syncthreads();
      v16h a0 = load_afrag(sP, brow, kbase, 0,  HH);
      v16h a1 = load_afrag(sP, brow, kbase, 32, HH);
      v8f c = {};
      c = wmma_f16f32(a0, bf0, c);
      c = wmma_f16f32(a1, bf1, c);
      float* bg = beta_g + (size_t)t * BB * HH;
#pragma unroll
      for (int r = 0; r < 8; ++r) {
        int b = ib * 16 + rbase + r;
        float v = sM[b] + __logf(c[r]);
        sA[b * HH + ncol] = v;
        bg[b * HH + ncol] = v;
      }
      __syncthreads();
    }
  }
}

// ---------------------------------------------------------------------------
// 64-thread block reductions
// ---------------------------------------------------------------------------
__device__ __forceinline__ float blockMax64(float v, float* red) {
  red[threadIdx.x] = v; __syncthreads();
  for (int s = 32; s > 0; s >>= 1) {
    if (threadIdx.x < s)
      red[threadIdx.x] = fmaxf(red[threadIdx.x], red[threadIdx.x + s]);
    __syncthreads();
  }
  float r = red[0]; __syncthreads();
  return r;
}
__device__ __forceinline__ float blockSum64(float v, float* red) {
  red[threadIdx.x] = v; __syncthreads();
  for (int s = 32; s > 0; s >>= 1) {
    if (threadIdx.x < s)
      red[threadIdx.x] = red[threadIdx.x] + red[threadIdx.x + s];
    __syncthreads();
  }
  float r = red[0]; __syncthreads();
  return r;
}

// ---------------------------------------------------------------------------
// K3: gamma-based terms (prior + emission), with per-sequence beta roll.
// One block per batch element, thread = hidden state.
// ---------------------------------------------------------------------------
__global__ void hmm_gamma_terms(const float* __restrict__ alpha_g,
                                const float* __restrict__ beta_g,
                                const float* __restrict__ lpx,
                                const float* __restrict__ logPriors,
                                const int* __restrict__ seqlen,
                                float* __restrict__ out) {
  __shared__ float red[64];
  const int b = blockIdx.x;
  const int h = threadIdx.x;
  const int L = seqlen[b];
  const float lp = logPriors[h];
  float acc = 0.f;
  for (int t = 0; t < TT; ++t) {
    int bt = t + TT - L; if (bt >= TT) bt -= TT;   // rolled beta index
    float g = alpha_g[(size_t)t * BB * HH + b * HH + h] +
              beta_g[(size_t)bt * BB * HH + b * HH + h];
    float mx = blockMax64(g, red);
    float e = __expf(g - mx);
    float s = blockSum64(e, red);
    float gamma = e / s;                            // exp(normalized log-gamma)
    if (t == 0) acc += gamma * lp;
    if (t < L)  acc += gamma * lpx[(size_t)t * BB * HH + b * HH + h];
  }
  float tot = blockSum64(acc, red);
  if (h == 0) atomicAdd(out, tot);
}

// ---------------------------------------------------------------------------
// K4: xi-based transition term. One block per (b, t in 0..T-2); thread = h.
// ---------------------------------------------------------------------------
__global__ void hmm_xi_terms(const float* __restrict__ alpha_g,
                             const float* __restrict__ beta_g,
                             const float* __restrict__ lpx,
                             const float* __restrict__ lT,
                             const int* __restrict__ seqlen,
                             float* __restrict__ out) {
  const int bt = blockIdx.x;                 // B*(T-1) blocks
  const int b = bt / (TT - 1), t = bt % (TT - 1);
  const int L = seqlen[b];
  if (t >= L - 1) return;                    // uniform per-block mask
  const int h = threadIdx.x;                 // 64 threads
  __shared__ float sy[64];
  __shared__ float red[64];
  int bt1 = (t + 1) + TT - L; if (bt1 >= TT) bt1 -= TT;
  sy[h] = lpx[(size_t)(t + 1) * BB * HH + b * HH + h] +
          beta_g[(size_t)bt1 * BB * HH + b * HH + h];
  __syncthreads();
  const float A = alpha_g[(size_t)t * BB * HH + b * HH + h];
  float mloc = NEG_INF;
  for (int hp = 0; hp < HH; ++hp)
    mloc = fmaxf(mloc, lT[h * HH + hp] + A + sy[hp]);
  float M = blockMax64(mloc, red);
  float s1 = 0.f, s2 = 0.f;
  for (int hp = 0; hp < HH; ++hp) {
    float lt = lT[h * HH + hp];
    float e = __expf(lt + A + sy[hp] - M);
    s1 += e; s2 += e * lt;
  }
  float S1 = blockSum64(s1, red);
  float S2 = blockSum64(s2, red);
  if (h == 0) atomicAdd(out, S2 / S1);
}

// ---------------------------------------------------------------------------
// Host launcher
// ---------------------------------------------------------------------------
extern "C" void kernel_launch(void* const* d_in, const int* in_sizes, int n_in,
                              void* d_out, int out_size, void* d_ws,
                              size_t ws_size, hipStream_t stream) {
  (void)in_sizes; (void)n_in; (void)out_size; (void)ws_size;
  const float* obs    = (const float*)d_in[0];
  const int*   seqlen = (const int*)  d_in[1];
  const float* sp     = (const float*)d_in[2];
  const float* utr    = (const float*)d_in[3];
  const float* uem    = (const float*)d_in[4];
  float* out = (float*)d_out;

  char* ws = (char*)d_ws;
  size_t off = 0;
  auto take = [&](size_t bytes) {
    char* p = ws + off;
    off = (off + bytes + 255) & ~(size_t)255;
    return p;
  };
  float*    logPriors = (float*)   take(HH * sizeof(float));
  float*    lT        = (float*)   take(HH * HH * sizeof(float));
  float*    lE        = (float*)   take(HH * OO * sizeof(float));
  _Float16* expT      = (_Float16*)take(HH * HH * sizeof(_Float16));
  _Float16* expTT     = (_Float16*)take(HH * HH * sizeof(_Float16));
  float*    lpx       = (float*)   take((size_t)TT * BB * HH * sizeof(float));
  float*    alpha     = (float*)   take((size_t)TT * BB * HH * sizeof(float));
  float*    beta      = (float*)   take((size_t)TT * BB * HH * sizeof(float));

  hmm_prep<<<1, 128, 0, stream>>>(sp, utr, uem, logPriors, lT, lE, expT, expTT, out);
  hmm_lpx<<<BB * TT, 64, 0, stream>>>(obs, lE, lpx);
  hmm_recurrence<<<2, 256, 0, stream>>>(lpx, logPriors, expT, expTT, alpha, beta);
  hmm_gamma_terms<<<BB, 64, 0, stream>>>(alpha, beta, lpx, logPriors, seqlen, out);
  hmm_xi_terms<<<BB * (TT - 1), 64, 0, stream>>>(alpha, beta, lpx, lT, seqlen, out);
}
